// GINRegressor_80968723464539
// MI455X (gfx1250) — compile-verified, matrix operand order checked by reading, not run on hardware
//
#include <hip/hip_runtime.h>
#include <hip/hip_bf16.h>
#include <stdint.h>
#include <stddef.h>

// ---------------------------------------------------------------------------
// Types for CDNA5 WMMA (wave32): A/B = 16 bf16 per lane, C/D = 8 f32 per lane
// ---------------------------------------------------------------------------
typedef __attribute__((ext_vector_type(16))) __bf16 v16bf;
typedef __attribute__((ext_vector_type(8)))  __bf16 v8bf;
typedef __attribute__((ext_vector_type(8)))  float  v8f;

#define HIDDEN 64

// ---------------------------------------------------------------------------
// A-fragment loader: 16x32 bf16 tile, per ISA 7.12.2 layout:
//   lanes 0-15 : M = lane,      VGPR0-3 K = [0..7],  VGPR4-7 K = [16..23]
//   lanes16-31 : M = lane-16,   VGPR0-3 K = [8..15], VGPR4-7 K = [24..31]
// Source is row-major f32; converts to bf16 on the fly.
// ---------------------------------------------------------------------------
__device__ __forceinline__ v16bf load_a_f32(const float* __restrict__ row,
                                            int kc, int khalf) {
  const float* q0 = row + kc + 8 * khalf;
  const float* q1 = row + kc + 16 + 8 * khalf;
  v16bf a;
#pragma unroll
  for (int j = 0; j < 8; ++j) a[j] = (__bf16)q0[j];
#pragma unroll
  for (int j = 0; j < 8; ++j) a[8 + j] = (__bf16)q1[j];
  return a;
}

// Packed B fragment: one aligned 32B vector per (chunk, col-tile, lane)
__device__ __forceinline__ v16bf load_b_pk(const __bf16* __restrict__ pk,
                                           int c, int wave, int lane) {
  return *(const v16bf*)(pk + ((size_t)((c * 4 + wave) * 32 + lane) << 4));
}

// ---------------------------------------------------------------------------
// Weight pre-swizzle: w is row-major [K][64] f32 -> bf16 fragments laid out so
// a GEMM lane reads its whole 16-element fragment contiguously.
//   out[((c*4 + t)*32 + lane)*16 + e] = bf16( w[k][t*16 + (lane&15)] )
//   k = 32c + (e<8 ? 8*(lane>>4)+e : 16 + 8*(lane>>4) + (e-8))
// ---------------------------------------------------------------------------
__global__ void pack_w_kernel(const float* __restrict__ w,
                              __bf16* __restrict__ out, int total) {
  int tid = blockIdx.x * blockDim.x + threadIdx.x;
  if (tid >= total) return;
  int e    = tid & 15;
  int lane = (tid >> 4) & 31;
  int ct   = tid >> 9;
  int t    = ct & 3;       // 4 column tiles (64 outputs)
  int c    = ct >> 2;      // K chunk
  int khalf = lane >> 4;
  int n = t * 16 + (lane & 15);
  int k = 32 * c + ((e < 8) ? (8 * khalf + e) : (16 + 8 * khalf + (e - 8)));
  out[tid] = (__bf16)w[(size_t)k * 64 + n];
}

// ---------------------------------------------------------------------------
// agg0 = x  (self term of first GIN layer), float4 copy
// ---------------------------------------------------------------------------
__global__ void copy_f32_kernel(const float* __restrict__ src,
                                float* __restrict__ dst, long long n4) {
  long long tid = (long long)blockIdx.x * blockDim.x + threadIdx.x;
  if (tid >= n4) return;
  ((float4*)dst)[tid] = ((const float4*)src)[tid];
}

__global__ void zero_f32_kernel(float* __restrict__ dst, long long n4) {
  long long tid = (long long)blockIdx.x * blockDim.x + threadIdx.x;
  if (tid >= n4) return;
  ((float4*)dst)[tid] = make_float4(0.f, 0.f, 0.f, 0.f);
}

// ---------------------------------------------------------------------------
// Edge scatter: agg[dst[e]] += h[src[e]]   (D = 4<<d4shift features)
// ---------------------------------------------------------------------------
__global__ void scatter_edges_kernel(const float* __restrict__ h,
                                     const long long* __restrict__ src,
                                     const long long* __restrict__ dst,
                                     float* __restrict__ agg,
                                     int d4shift, long long total) {
  long long tid = (long long)blockIdx.x * blockDim.x + threadIdx.x;
  if (tid >= total) return;
  long long e = tid >> d4shift;
  int g = (int)(tid & ((1 << d4shift) - 1));
  int D = 4 << d4shift;
  long long s = src[e], d = dst[e];
  float4 v = *(const float4*)(h + (size_t)s * D + g * 4);
  float* p = agg + (size_t)d * D + g * 4;
  atomicAdd(p + 0, v.x);
  atomicAdd(p + 1, v.y);
  atomicAdd(p + 2, v.z);
  atomicAdd(p + 3, v.w);
}

// ---------------------------------------------------------------------------
// Fused GIN MLP: h = relu( relu(agg@wa+ba) @ wb + bb ); agg_next = h
// One 16-row tile per workgroup; 4 waves, wave w owns output cols [16w,16w+16).
// Both GEMMs via v_wmma_f32_16x16x32_bf16, z1 staged in LDS as bf16.
// ---------------------------------------------------------------------------
template <int DIN>
__global__ __launch_bounds__(128)
void gin_mlp_kernel(const float* __restrict__ agg,
                    const __bf16* __restrict__ wa_pk, const float* __restrict__ ba,
                    const __bf16* __restrict__ wb_pk, const float* __restrict__ bb,
                    float* __restrict__ h_out, float* __restrict__ agg_next,
                    int nrows) {
  __shared__ __bf16 z1[16][HIDDEN];

  const int lane  = threadIdx.x & 31;
  const int wave  = threadIdx.x >> 5;     // column tile
  const int m     = lane & 15;
  const int khalf = lane >> 4;
  const int n     = wave * 16 + m;        // global output column
  const int row0  = blockIdx.x * 16;

  int arow = row0 + m;
  if (arow >= nrows) arow = nrows - 1;    // clamp (stores are guarded)
  const float* ap = agg + (size_t)arow * DIN;

  // ---- GEMM 1: z1 = relu(agg @ wa + ba) ----
  v8f acc = {0.f, 0.f, 0.f, 0.f, 0.f, 0.f, 0.f, 0.f};
#pragma unroll
  for (int c = 0; c < DIN / 32; ++c) {
    v16bf a = load_a_f32(ap, 32 * c, khalf);
    v16bf b = load_b_pk(wa_pk, c, wave, lane);
    acc = __builtin_amdgcn_wmma_f32_16x16x32_bf16(false, a, false, b,
                                                  (short)0, acc, false, false);
  }
  {
    float bias = ba[n];
#pragma unroll
    for (int r = 0; r < 8; ++r) {
      float z = acc[r] + bias;
      z = z > 0.f ? z : 0.f;
      z1[r + 8 * khalf][n] = (__bf16)z;   // D layout: M = r + 8*(lane>>4)
    }
  }
  __syncthreads();

  // ---- GEMM 2: h = relu(z1 @ wb + bb) ----
  v8f acc2 = {0.f, 0.f, 0.f, 0.f, 0.f, 0.f, 0.f, 0.f};
#pragma unroll
  for (int c = 0; c < HIDDEN / 32; ++c) {
    const __bf16* zr = &z1[m][32 * c];
    v8bf lo = *(const v8bf*)(zr + 8 * khalf);
    v8bf hi = *(const v8bf*)(zr + 16 + 8 * khalf);
    v16bf a;
#pragma unroll
    for (int j = 0; j < 8; ++j) { a[j] = lo[j]; a[8 + j] = hi[j]; }
    v16bf b = load_b_pk(wb_pk, c, wave, lane);
    acc2 = __builtin_amdgcn_wmma_f32_16x16x32_bf16(false, a, false, b,
                                                   (short)0, acc2, false, false);
  }
  {
    float bias = bb[n];
#pragma unroll
    for (int r = 0; r < 8; ++r) {
      int rr = row0 + r + 8 * khalf;
      if (rr < nrows) {
        float v = acc2[r] + bias;
        v = v > 0.f ? v : 0.f;
        size_t idx = (size_t)rr * HIDDEN + n;
        h_out[idx] = v;
        if (agg_next) agg_next[idx] = v;   // next layer's self term
      }
    }
  }
}

// ---------------------------------------------------------------------------
// Global-add-pool, exploiting sorted `batch`: each thread owns one float4
// feature group over a 32-node chunk, accumulates runs with equal graph id in
// registers, and flushes one atomic per run boundary (~16x fewer atomics).
// ---------------------------------------------------------------------------
__global__ void pool_seg_kernel(const float* __restrict__ h,
                                const long long* __restrict__ batch,
                                float* __restrict__ pooled,
                                long long nnodes, long long nthreads) {
  long long tid = (long long)blockIdx.x * blockDim.x + threadIdx.x;
  if (tid >= nthreads) return;
  const int g = (int)(tid & 15);          // float4 feature group
  long long i0 = (tid >> 4) * 32;
  if (i0 >= nnodes) return;
  long long iend = i0 + 32;
  if (iend > nnodes) iend = nnodes;

  long long cur = batch[i0];
  float4 acc = make_float4(0.f, 0.f, 0.f, 0.f);
  for (long long i = i0; i < iend; ++i) {
    long long b = batch[i];
    if (b != cur) {
      float* p = pooled + (size_t)cur * HIDDEN + g * 4;
      atomicAdd(p + 0, acc.x);
      atomicAdd(p + 1, acc.y);
      atomicAdd(p + 2, acc.z);
      atomicAdd(p + 3, acc.w);
      acc = make_float4(0.f, 0.f, 0.f, 0.f);
      cur = b;
    }
    float4 v = *(const float4*)(h + (size_t)i * HIDDEN + g * 4);
    acc.x += v.x; acc.y += v.y; acc.z += v.z; acc.w += v.w;
  }
  float* p = pooled + (size_t)cur * HIDDEN + g * 4;
  atomicAdd(p + 0, acc.x);
  atomicAdd(p + 1, acc.y);
  atomicAdd(p + 2, acc.z);
  atomicAdd(p + 3, acc.w);
}

// ---------------------------------------------------------------------------
// Head: out = relu(pooled @ wh1 + bh1) @ wh2 + bh2   -> [G]
// GEMM1 via WMMA; second linear is a 64-wide row dot reduced through LDS.
// ---------------------------------------------------------------------------
__global__ __launch_bounds__(128)
void head_kernel(const float* __restrict__ pooled,
                 const __bf16* __restrict__ wh1_pk, const float* __restrict__ bh1,
                 const float* __restrict__ wh2, const float* __restrict__ bh2,
                 float* __restrict__ out, int ngraphs) {
  __shared__ float rowsum[16];
  const int lane  = threadIdx.x & 31;
  const int wave  = threadIdx.x >> 5;
  const int m     = lane & 15;
  const int khalf = lane >> 4;
  const int n     = wave * 16 + m;
  const int g0    = blockIdx.x * 16;

  if (threadIdx.x < 16) rowsum[threadIdx.x] = 0.f;
  __syncthreads();

  int grow = g0 + m;
  if (grow >= ngraphs) grow = ngraphs - 1;
  const float* pp = pooled + (size_t)grow * HIDDEN;

  v8f acc = {0.f, 0.f, 0.f, 0.f, 0.f, 0.f, 0.f, 0.f};
#pragma unroll
  for (int c = 0; c < HIDDEN / 32; ++c) {
    v16bf a = load_a_f32(pp, 32 * c, khalf);
    v16bf b = load_b_pk(wh1_pk, c, wave, lane);
    acc = __builtin_amdgcn_wmma_f32_16x16x32_bf16(false, a, false, b,
                                                  (short)0, acc, false, false);
  }
  float bias = bh1[n];
  float w2   = wh2[n];
#pragma unroll
  for (int r = 0; r < 8; ++r) {
    float z = acc[r] + bias;
    z = z > 0.f ? z : 0.f;
    atomicAdd(&rowsum[r + 8 * khalf], z * w2);
  }
  __syncthreads();

  if (threadIdx.x < 16 && (g0 + threadIdx.x) < ngraphs)
    out[g0 + threadIdx.x] = rowsum[threadIdx.x] + bh2[0];
}

// ---------------------------------------------------------------------------
// Host orchestration
// ---------------------------------------------------------------------------
static inline size_t align256(size_t x) { return (x + 255) & ~(size_t)255; }

extern "C" void kernel_launch(void* const* d_in, const int* in_sizes, int n_in,
                              void* d_out, int out_size, void* d_ws, size_t ws_size,
                              hipStream_t stream) {
  (void)n_in; (void)ws_size;

  const int IN_DIM = 128;
  const long long N = in_sizes[0] / IN_DIM;
  const long long E = in_sizes[1] / 2;
  const int G = out_size;

  const float*     x    = (const float*)d_in[0];
  const long long* ei   = (const long long*)d_in[1];
  const long long* src  = ei;
  const long long* dst  = ei + E;
  const long long* batch = (const long long*)d_in[2];
  const float* w0a = (const float*)d_in[3];  const float* b0a = (const float*)d_in[4];
  const float* w0b = (const float*)d_in[5];  const float* b0b = (const float*)d_in[6];
  const float* w1a = (const float*)d_in[7];  const float* b1a = (const float*)d_in[8];
  const float* w1b = (const float*)d_in[9];  const float* b1b = (const float*)d_in[10];
  const float* w2a = (const float*)d_in[11]; const float* b2a = (const float*)d_in[12];
  const float* w2b = (const float*)d_in[13]; const float* b2b = (const float*)d_in[14];
  const float* wh1 = (const float*)d_in[15]; const float* bh1 = (const float*)d_in[16];
  const float* wh2 = (const float*)d_in[17]; const float* bh2 = (const float*)d_in[18];
  float* out = (float*)d_out;

  // ---- workspace layout ----
  char* ws = (char*)d_ws;
  size_t off = 0;
  float* aggA = (float*)(ws + off); off = align256(off + (size_t)N * IN_DIM * 4); // N x 128 (reused N x 64)
  float* hbuf = (float*)(ws + off); off = align256(off + (size_t)N * HIDDEN * 4); // N x 64
  float* aggB = (float*)(ws + off); off = align256(off + (size_t)N * HIDDEN * 4); // N x 64
  float* pooled = (float*)(ws + off); off = align256(off + (size_t)G * HIDDEN * 4);
  const int PK128 = (IN_DIM / 32) * 4 * 32 * 16;  // 8192 bf16
  const int PK64  = (HIDDEN / 32) * 4 * 32 * 16;  // 4096 bf16
  __bf16* pk_w0a = (__bf16*)(ws + off); off = align256(off + (size_t)PK128 * 2);
  __bf16* pk_w0b = (__bf16*)(ws + off); off = align256(off + (size_t)PK64 * 2);
  __bf16* pk_w1a = (__bf16*)(ws + off); off = align256(off + (size_t)PK64 * 2);
  __bf16* pk_w1b = (__bf16*)(ws + off); off = align256(off + (size_t)PK64 * 2);
  __bf16* pk_w2a = (__bf16*)(ws + off); off = align256(off + (size_t)PK64 * 2);
  __bf16* pk_w2b = (__bf16*)(ws + off); off = align256(off + (size_t)PK64 * 2);
  __bf16* pk_wh1 = (__bf16*)(ws + off); off = align256(off + (size_t)PK64 * 2);

  const int TB = 256;

  // ---- 0. pack weights into WMMA B-fragment order ----
  pack_w_kernel<<<(PK128 + TB - 1) / TB, TB, 0, stream>>>(w0a, pk_w0a, PK128);
  pack_w_kernel<<<(PK64 + TB - 1) / TB, TB, 0, stream>>>(w0b, pk_w0b, PK64);
  pack_w_kernel<<<(PK64 + TB - 1) / TB, TB, 0, stream>>>(w1a, pk_w1a, PK64);
  pack_w_kernel<<<(PK64 + TB - 1) / TB, TB, 0, stream>>>(w1b, pk_w1b, PK64);
  pack_w_kernel<<<(PK64 + TB - 1) / TB, TB, 0, stream>>>(w2a, pk_w2a, PK64);
  pack_w_kernel<<<(PK64 + TB - 1) / TB, TB, 0, stream>>>(w2b, pk_w2b, PK64);
  pack_w_kernel<<<(PK64 + TB - 1) / TB, TB, 0, stream>>>(wh1, pk_wh1, PK64);

  const int nTiles = (int)((N + 15) / 16);
  const long long e128 = E * (IN_DIM / 4);   // E * 32
  const long long e64  = E * (HIDDEN / 4);   // E * 16

  // ---- layer 0 ----
  {
    long long n4 = N * IN_DIM / 4;
    copy_f32_kernel<<<(unsigned)((n4 + TB - 1) / TB), TB, 0, stream>>>(x, aggA, n4);
  }
  scatter_edges_kernel<<<(unsigned)((e128 + TB - 1) / TB), TB, 0, stream>>>(
      x, src, dst, aggA, 5, e128);
  gin_mlp_kernel<128><<<nTiles, 128, 0, stream>>>(
      aggA, pk_w0a, b0a, pk_w0b, b0b, hbuf, aggB, (int)N);

  // ---- layer 1 ----
  scatter_edges_kernel<<<(unsigned)((e64 + TB - 1) / TB), TB, 0, stream>>>(
      hbuf, src, dst, aggB, 4, e64);
  gin_mlp_kernel<64><<<nTiles, 128, 0, stream>>>(
      aggB, pk_w1a, b1a, pk_w1b, b1b, hbuf, aggA, (int)N);  // aggA reused as N x 64

  // ---- layer 2 ----
  scatter_edges_kernel<<<(unsigned)((e64 + TB - 1) / TB), TB, 0, stream>>>(
      hbuf, src, dst, aggA, 4, e64);
  gin_mlp_kernel<64><<<nTiles, 128, 0, stream>>>(
      aggA, pk_w2a, b2a, pk_w2b, b2b, hbuf, (float*)nullptr, (int)N);

  // ---- pool + head ----
  {
    long long p4 = (long long)G * HIDDEN / 4;
    zero_f32_kernel<<<(unsigned)((p4 + TB - 1) / TB), TB, 0, stream>>>(pooled, p4);
  }
  {
    long long nthreads = ((N + 31) / 32) * 16;
    pool_seg_kernel<<<(unsigned)((nthreads + TB - 1) / TB), TB, 0, stream>>>(
        hbuf, batch, pooled, N, nthreads);
  }
  head_kernel<<<(G + 15) / 16, 128, 0, stream>>>(
      pooled, pk_wh1, bh1, wh2, bh2, out, G);
}
